// GraphProcessor_14164802142586
// MI455X (gfx1250) — matrix operand order repaired and checked by reading, not compiled, not run on hardware
//
#include <hip/hip_runtime.h>
#include <math.h>

#define DIMF   128
#define HID    512
#define NLAYERS 2
#define NLEGO  50000
#define NPOINT 100000
#define ELL    500000
#define EPP    400000
#define ELP    500000
#define EPL    500000

typedef __attribute__((ext_vector_type(16))) _Float16 v16h;
typedef __attribute__((ext_vector_type(8)))  _Float16 v8h;
typedef __attribute__((ext_vector_type(8)))  float    v8f;

// ---------------- elementwise helpers ----------------

__global__ void k_fill(float* __restrict__ p, float v, long long n) {
  long long i = (long long)blockIdx.x * blockDim.x + threadIdx.x;
  if (i < n) p[i] = v;
}

__global__ void k_cvt16(const float* __restrict__ in, _Float16* __restrict__ out, long long n) {
  long long i = (long long)blockIdx.x * blockDim.x + threadIdx.x;
  if (i < n) out[i] = (_Float16)in[i];
}

__global__ void k_addrow(float* __restrict__ out, const float* __restrict__ bias, long long n) {
  long long i = (long long)blockIdx.x * blockDim.x + threadIdx.x;
  if (i < n) out[i] += bias[i & (DIMF - 1)];
}

// out = (isfinite(mx) ? mx : 0) + bias[col]
__global__ void k_finmax(const float* __restrict__ mx, const float* __restrict__ bias,
                         float* __restrict__ out, long long n) {
  long long i = (long long)blockIdx.x * blockDim.x + threadIdx.x;
  if (i >= n) return;
  float v = mx[i];
  if (!__builtin_isfinite(v)) v = 0.0f;
  if (bias) v += bias[i & (DIMF - 1)];
  out[i] = v;
}

// ---------------- weight packing into WMMA B-operand layout ----------------
// Bp[tile=kt*(N/16)+nt][lane] : v16h, element e -> B[kt*32 + (lane>>4)*16 + e][nt*16 + (lane&15)]
// optional Bsub: pack (B - Bsub) (used for EdgeConv W1_top - W1_bot)
__global__ void k_pack(const float* __restrict__ B, const float* __restrict__ Bsub,
                       v16h* __restrict__ Bp, int K, int N) {
  int t = blockIdx.x * blockDim.x + threadIdx.x;
  int lane = t & 31, tile = t >> 5;
  int tilesN = N >> 4;
  if (tile >= (K >> 5) * tilesN) return;
  int kt = tile / tilesN, nt = tile - kt * tilesN;
  int col = nt * 16 + (lane & 15);
  int kbase = kt * 32 + (lane >> 4) * 16;
  v16h v;
#pragma unroll
  for (int e = 0; e < 16; ++e) {
    float x = B[(size_t)(kbase + e) * N + col];
    if (Bsub) x -= Bsub[(size_t)(kbase + e) * N + col];
    v[e] = (_Float16)x;
  }
  Bp[(size_t)tile * 32 + lane] = v;
}

// ---------------- generic WMMA GEMM: C[MxN](f32 or f16) = A16[MxK] * Bp + bias ----------------
// one wave per 16x16 output tile; K multiple of 32; M,N multiples of 16
template <bool OUT_HALF>
__global__ void k_gemm(const _Float16* __restrict__ A, const v16h* __restrict__ Bp,
                       const float* __restrict__ bias, void* __restrict__ C,
                       int M, int N, int K) {
  int lane = threadIdx.x & 31;
  int wave = blockIdx.x * (blockDim.x >> 5) + (threadIdx.x >> 5);
  int tilesN = N >> 4;
  int tileM = wave / tilesN, tileN = wave - tileM * tilesN;
  if (tileM >= (M >> 4)) return;
  int row = lane & 15, hi = lane >> 4;
  const _Float16* arow = A + (size_t)(tileM * 16 + row) * K;
  v8f c = {};
  int ks_n = K >> 5;
  for (int ks = 0; ks < ks_n; ++ks) {
    union { v16h v; v8h h[2]; } ua;
    // A 16x32 f16, contiguous k order used here (same as k_pack): lane(row,hi): elems 0..15 -> K = hi*16+j
    ua.h[0] = *(const v8h*)(arow + ks * 32 + hi * 16);
    ua.h[1] = *(const v8h*)(arow + ks * 32 + hi * 16 + 8);
    v16h b = Bp[((size_t)ks * tilesN + tileN) * 32 + lane];
    c = __builtin_amdgcn_wmma_f32_16x16x32_f16(false, ua.v, false, b, (short)0, c, false, false);
  }
  int col = tileN * 16 + (lane & 15);
  float bv = bias ? bias[col] : 0.0f;
#pragma unroll
  for (int r = 0; r < 8; ++r) {
    int m = tileM * 16 + r + 8 * hi;   // C/D layout: VGPR r holds M = r + 8*hi
    float v = c[r] + bv;
    if (OUT_HALF) ((_Float16*)C)[(size_t)m * N + col] = (_Float16)v;
    else          ((float*)C)[(size_t)m * N + col] = v;
  }
}

// ---------------- edge-level kernels ----------------

__device__ __forceinline__ float wave_sum(float s) {
#pragma unroll
  for (int off = 16; off > 0; off >>= 1) s += __shfl_down(s, off, 32);
  return s;
}

// TransformerConv score: dot(Q[dst], K[src]) / sqrt(128); one wave per edge
__global__ void k_trans_score(const float* __restrict__ Q, const float* __restrict__ Kf,
                              const int* __restrict__ src, const int* __restrict__ dst,
                              float* __restrict__ score, int E) {
  int lane = threadIdx.x & 31;
  int e = (blockIdx.x * blockDim.x + threadIdx.x) >> 5;
  if (e >= E) return;
  const float* q = Q + (size_t)dst[e] * DIMF;
  const float* k = Kf + (size_t)src[e] * DIMF;
  float s = 0.0f;
#pragma unroll
  for (int j = lane; j < DIMF; j += 32) s = fmaf(q[j], k[j], s);
  s = wave_sum(s);
  if (lane == 0) score[e] = s * 0.08838834764831845f; // 1/sqrt(128)
}

// per-node dot with attention vector (GAT alpha); one wave per node
__global__ void k_node_dot(const float* __restrict__ H, const float* __restrict__ a,
                           float* __restrict__ out, int n) {
  int lane = threadIdx.x & 31;
  int i = (blockIdx.x * blockDim.x + threadIdx.x) >> 5;
  if (i >= n) return;
  float s = 0.0f;
#pragma unroll
  for (int j = lane; j < DIMF; j += 32) s = fmaf(H[(size_t)i * DIMF + j], a[j], s);
  s = wave_sum(s);
  if (lane == 0) out[i] = s;
}

__global__ void k_gat_score(const float* __restrict__ as, const float* __restrict__ ad,
                            const int* __restrict__ src, const int* __restrict__ dst,
                            float* __restrict__ score, int E) {
  int e = blockIdx.x * blockDim.x + threadIdx.x;
  if (e >= E) return;
  float s = as[src[e]] + ad[dst[e]];
  score[e] = (s > 0.0f) ? s : 0.2f * s;  // leaky_relu 0.2
}

__global__ void k_segmax(const float* __restrict__ score, const int* __restrict__ dst,
                         float* __restrict__ m, int E) {
  int e = blockIdx.x * blockDim.x + threadIdx.x;
  if (e >= E) return;
  __hip_atomic_fetch_max(&m[dst[e]], score[e], __ATOMIC_RELAXED, __HIP_MEMORY_SCOPE_AGENT);
}

__global__ void k_finm(float* __restrict__ m, int n) {
  int i = blockIdx.x * blockDim.x + threadIdx.x;
  if (i >= n) return;
  float v = m[i];
  if (!__builtin_isfinite(v)) m[i] = 0.0f;
}

__global__ void k_expden(float* __restrict__ score, const int* __restrict__ dst,
                         const float* __restrict__ m, float* __restrict__ denom, int E) {
  int e = blockIdx.x * blockDim.x + threadIdx.x;
  if (e >= E) return;
  int d = dst[e];
  float ev = __expf(score[e] - m[d]);
  score[e] = ev;
  atomicAdd(&denom[d], ev);
}

// out[dst] += (score/denom[dst]) * V[src]; one wave per edge
__global__ void k_aggregate(const float* __restrict__ score, const float* __restrict__ denom,
                            const float* __restrict__ V, const int* __restrict__ src,
                            const int* __restrict__ dst, float* __restrict__ out, int E) {
  int lane = threadIdx.x & 31;
  int e = (blockIdx.x * blockDim.x + threadIdx.x) >> 5;
  if (e >= E) return;
  int s = src[e], d = dst[e];
  float a = score[e] / (denom[d] + 1e-16f);
  const float* v = V + (size_t)s * DIMF;
  float* o = out + (size_t)d * DIMF;
#pragma unroll
  for (int j = lane; j < DIMF; j += 32) atomicAdd(&o[j], a * v[j]);
}

// ---------------- fused EdgeConv edge-GEMM (the WMMA hot loop) ----------------
// block = 256 threads = 8 waves; 16 edges per block; K=512, N=128 (8 n-tiles, one per wave)
// hs[e][k] = relu(D[dst_e][k] + Q[src_e][k]); m = hs @ W2 + b2; atomicMax into out[dst_e]
__global__ void __launch_bounds__(256)
k_edgeconv(const _Float16* __restrict__ D, const _Float16* __restrict__ Q,
           const v16h* __restrict__ Bp2, const float* __restrict__ b2,
           const int* __restrict__ src, const int* __restrict__ dst,
           float* __restrict__ out, int E) {
  __shared__ _Float16 hs[16][HID];
  __shared__ int sdst[16];
  int e0 = blockIdx.x * 16;
  for (int idx = threadIdx.x; idx < 16 * HID; idx += 256) {
    int e = idx >> 9, k = idx & (HID - 1);
    int s = src[e0 + e], d = dst[e0 + e];
    float h = (float)D[(size_t)d * HID + k] + (float)Q[(size_t)s * HID + k];
    hs[e][k] = (_Float16)(h > 0.0f ? h : 0.0f);
  }
  if (threadIdx.x < 16) sdst[threadIdx.x] = dst[e0 + threadIdx.x];
  __syncthreads();

  int w = threadIdx.x >> 5;        // n-tile 0..7
  int lane = threadIdx.x & 31;
  int row = lane & 15, hi = lane >> 4;
  v8f c = {};
#pragma unroll
  for (int ks = 0; ks < HID / 32; ++ks) {
    union { v16h v; v8h h[2]; } ua;
    ua.h[0] = *(const v8h*)&hs[row][ks * 32 + hi * 16];
    ua.h[1] = *(const v8h*)&hs[row][ks * 32 + hi * 16 + 8];
    v16h b = Bp2[(size_t)(ks * 8 + w) * 32 + lane];
    c = __builtin_amdgcn_wmma_f32_16x16x32_f16(false, ua.v, false, b, (short)0, c, false, false);
  }
  int col = w * 16 + (lane & 15);
  float bias = b2[col];
#pragma unroll
  for (int r = 0; r < 8; ++r) {
    int e = r + 8 * hi;
    float v = c[r] + bias;
    __hip_atomic_fetch_max(&out[(size_t)sdst[e] * DIMF + col], v,
                           __ATOMIC_RELAXED, __HIP_MEMORY_SCOPE_AGENT);
  }
}

// ---------------- host-side launch helpers ----------------

static inline void run_fill(hipStream_t st, float* p, float v, long long n) {
  k_fill<<<dim3((unsigned)((n + 255) / 256)), dim3(256), 0, st>>>(p, v, n);
}
static inline void run_cvt(hipStream_t st, const float* in, _Float16* out, long long n) {
  k_cvt16<<<dim3((unsigned)((n + 255) / 256)), dim3(256), 0, st>>>(in, out, n);
}
static inline void run_pack(hipStream_t st, const float* W, const float* Wsub, v16h* Bp, int K, int N) {
  int th = (K / 32) * (N / 16) * 32;
  k_pack<<<dim3((th + 255) / 256), dim3(256), 0, st>>>(W, Wsub, Bp, K, N);
}
static inline void run_gemm(hipStream_t st, const _Float16* A, const v16h* Bp, const float* bias,
                            void* C, int M, int N, int K, bool half_out) {
  long long tiles = (long long)(M / 16) * (N / 16);
  unsigned blocks = (unsigned)((tiles + 7) / 8);
  if (half_out) k_gemm<true><<<dim3(blocks), dim3(256), 0, st>>>(A, Bp, bias, C, M, N, K);
  else          k_gemm<false><<<dim3(blocks), dim3(256), 0, st>>>(A, Bp, bias, C, M, N, K);
}
static inline void run_softmax(hipStream_t st, float* score, const int* edst, int E, int Nn,
                               float* mb, float* db) {
  run_fill(st, mb, -INFINITY, Nn);
  k_segmax<<<dim3((E + 255) / 256), dim3(256), 0, st>>>(score, edst, mb, E);
  k_finm<<<dim3((Nn + 255) / 256), dim3(256), 0, st>>>(mb, Nn);
  run_fill(st, db, 0.0f, Nn);
  k_expden<<<dim3((E + 255) / 256), dim3(256), 0, st>>>(score, edst, mb, db, E);
}
static inline void run_aggregate(hipStream_t st, const float* score, const float* db,
                                 const float* V, const int* esrc, const int* edst,
                                 float* out, int E) {
  k_aggregate<<<dim3((E + 7) / 8), dim3(256), 0, st>>>(score, db, V, esrc, edst, out, E);
}

extern "C" void kernel_launch(void* const* d_in, const int* in_sizes, int n_in,
                              void* d_out, int out_size, void* d_ws, size_t ws_size,
                              hipStream_t stream) {
  const float* x_lego  = (const float*)d_in[0];
  const float* x_point = (const float*)d_in[1];
  const float* tWq = (const float*)d_in[2];
  const float* tbq = (const float*)d_in[3];
  const float* tWk = (const float*)d_in[4];
  const float* tbk = (const float*)d_in[5];
  const float* tWv = (const float*)d_in[6];
  const float* tbv = (const float*)d_in[7];
  const float* tWs = (const float*)d_in[8];
  const float* tbs = (const float*)d_in[9];
  const float* eW1 = (const float*)d_in[10];
  const float* eb1 = (const float*)d_in[11];
  const float* eW2 = (const float*)d_in[12];
  const float* eb2 = (const float*)d_in[13];
  const float* gW  = (const float*)d_in[14];
  const float* gAs = (const float*)d_in[15];
  const float* gAd = (const float*)d_in[16];
  const float* gb  = (const float*)d_in[17];
  const int* ll_src = (const int*)d_in[18];
  const int* ll_dst = (const int*)d_in[19];
  const int* pp_src = (const int*)d_in[20];
  const int* pp_dst = (const int*)d_in[21];
  const int* lp_src = (const int*)d_in[22];
  const int* lp_dst = (const int*)d_in[23];
  const int* pl_src = (const int*)d_in[24];
  const int* pl_dst = (const int*)d_in[25];

  float* out_lego  = (float*)d_out;
  float* out_point = (float*)d_out + (size_t)NLEGO * DIMF;

  // ---- workspace carving ----
  char* ws = (char*)d_ws;
  size_t off = 0;
  auto alloc = [&](size_t bytes) -> void* {
    off = (off + 255) & ~(size_t)255;
    void* p = ws + off; off += bytes; return p;
  };
  _Float16* xl16 = (_Float16*)alloc((size_t)NLEGO  * DIMF * 2);
  _Float16* xp16 = (_Float16*)alloc((size_t)NPOINT * DIMF * 2);
  // U1: TransConv Q|K|V (3 x 50k x 128 f32)  OR  EdgeConv D (100k x 512 f16) -- never live together
  char* U1 = (char*)alloc((size_t)NPOINT * HID * 2);
  // U2: GAT Hs(100k)+Hd(100k) f32  OR  EdgeConv Qsrc (100k x 512 f16)
  char* U2 = (char*)alloc((size_t)NPOINT * HID * 2);
  float* HsL    = (float*)alloc((size_t)NLEGO * DIMF * 4);   // GAT lego->point Hs (kept across EdgeConv)
  float* maxbuf = (float*)alloc((size_t)NPOINT * DIMF * 4);
  float* lego_a  = (float*)alloc((size_t)NLEGO  * DIMF * 4);
  float* point_a = (float*)alloc((size_t)NPOINT * DIMF * 4);
  float* alpha_s = (float*)alloc((size_t)NPOINT * 4);
  float* alpha_d = (float*)alloc((size_t)NPOINT * 4);
  float* score   = (float*)alloc((size_t)ELL * 4);
  float* mb      = (float*)alloc((size_t)NPOINT * 4);
  float* db      = (float*)alloc((size_t)NPOINT * 4);
  v16h* BpS = (v16h*)alloc((size_t)DIMF * DIMF * 2);   // 128x128 pack
  v16h* BpL = (v16h*)alloc((size_t)DIMF * HID * 2);    // 128x512 pack
  v16h* Bp2 = (v16h*)alloc((size_t)HID * DIMF * 2);    // 512x128 pack

  float* bufQ = (float*)U1;
  float* bufK = (float*)(U1 + (size_t)NLEGO * DIMF * 4);
  float* bufV = (float*)(U1 + (size_t)NLEGO * DIMF * 8);
  _Float16* Dbuf = (_Float16*)U1;
  float* Hs_pl = (float*)U2;                                // point-side H (100k)
  float* Hd_pl = (float*)(U2 + (size_t)NPOINT * DIMF * 4);  // lego-side H
  float* Hd_lp = (float*)U2;                                // point-side H for lp
  _Float16* Qbuf = (_Float16*)U2;

  const float* curL = x_lego;
  const float* curP = x_point;

  for (int layer = 0; layer < NLAYERS; ++layer) {
    const int iA = 2 * layer, iB = 2 * layer + 1;
    const int i_lp = 2 * layer, i_pl = 2 * layer + 1;

    // ---- f16 feature copies ----
    run_cvt(stream, curL, xl16, (long long)NLEGO * DIMF);
    run_cvt(stream, curP, xp16, (long long)NPOINT * DIMF);

    // ================= conv A =================
    // --- TransformerConv lego->lego (instance iA) ---
    run_pack(stream, tWq + (size_t)iA * DIMF * DIMF, nullptr, BpS, DIMF, DIMF);
    run_gemm(stream, xl16, BpS, tbq + (size_t)iA * DIMF, bufQ, NLEGO, DIMF, DIMF, false);
    run_pack(stream, tWk + (size_t)iA * DIMF * DIMF, nullptr, BpS, DIMF, DIMF);
    run_gemm(stream, xl16, BpS, tbk + (size_t)iA * DIMF, bufK, NLEGO, DIMF, DIMF, false);
    run_pack(stream, tWv + (size_t)iA * DIMF * DIMF, nullptr, BpS, DIMF, DIMF);
    run_gemm(stream, xl16, BpS, tbv + (size_t)iA * DIMF, bufV, NLEGO, DIMF, DIMF, false);
    run_pack(stream, tWs + (size_t)iA * DIMF * DIMF, nullptr, BpS, DIMF, DIMF);
    run_gemm(stream, xl16, BpS, tbs + (size_t)iA * DIMF, lego_a, NLEGO, DIMF, DIMF, false); // skip term
    k_trans_score<<<dim3((ELL + 7) / 8), dim3(256), 0, stream>>>(bufQ, bufK, ll_src, ll_dst, score, ELL);
    run_softmax(stream, score, ll_dst, ELL, NLEGO, mb, db);
    run_aggregate(stream, score, db, bufV, ll_src, ll_dst, lego_a, ELL);
    k_addrow<<<dim3((unsigned)(((long long)NLEGO * DIMF + 255) / 256)), dim3(256), 0, stream>>>(
        lego_a, gb + (size_t)i_pl * DIMF, (long long)NLEGO * DIMF);  // GAT pl bias

    // --- GAT point->lego (instance i_pl) ---
    run_pack(stream, gW + (size_t)i_pl * DIMF * DIMF, nullptr, BpS, DIMF, DIMF);
    run_gemm(stream, xp16, BpS, nullptr, Hs_pl, NPOINT, DIMF, DIMF, false);
    run_gemm(stream, xl16, BpS, nullptr, Hd_pl, NLEGO, DIMF, DIMF, false);
    k_node_dot<<<dim3((NPOINT * 32 + 255) / 256), dim3(256), 0, stream>>>(Hs_pl, gAs + (size_t)i_pl * DIMF, alpha_s, NPOINT);
    k_node_dot<<<dim3((NLEGO * 32 + 255) / 256), dim3(256), 0, stream>>>(Hd_pl, gAd + (size_t)i_pl * DIMF, alpha_d, NLEGO);
    k_gat_score<<<dim3((EPL + 255) / 256), dim3(256), 0, stream>>>(alpha_s, alpha_d, pl_src, pl_dst, score, EPL);
    run_softmax(stream, score, pl_dst, EPL, NLEGO, mb, db);
    run_aggregate(stream, score, db, Hs_pl, pl_src, pl_dst, lego_a, EPL);

    // --- GAT lego->point (instance i_lp): scores+softmax now, aggregate after EdgeConv ---
    run_pack(stream, gW + (size_t)i_lp * DIMF * DIMF, nullptr, BpS, DIMF, DIMF);
    run_gemm(stream, xl16, BpS, nullptr, HsL, NLEGO, DIMF, DIMF, false);
    run_gemm(stream, xp16, BpS, nullptr, Hd_lp, NPOINT, DIMF, DIMF, false);
    k_node_dot<<<dim3((NLEGO * 32 + 255) / 256), dim3(256), 0, stream>>>(HsL, gAs + (size_t)i_lp * DIMF, alpha_s, NLEGO);
    k_node_dot<<<dim3((NPOINT * 32 + 255) / 256), dim3(256), 0, stream>>>(Hd_lp, gAd + (size_t)i_lp * DIMF, alpha_d, NPOINT);
    k_gat_score<<<dim3((ELP + 255) / 256), dim3(256), 0, stream>>>(alpha_s, alpha_d, lp_src, lp_dst, score, ELP);
    run_softmax(stream, score, lp_dst, ELP, NPOINT, mb, db);

    // --- EdgeConv point->point (instance iA); overwrites U1(D), U2(Q) ---
    run_pack(stream, eW1 + (size_t)iA * 2 * DIMF * HID,
             eW1 + (size_t)iA * 2 * DIMF * HID + (size_t)DIMF * HID, BpL, DIMF, HID); // W1top - W1bot
    run_gemm(stream, xp16, BpL, eb1 + (size_t)iA * HID, Dbuf, NPOINT, HID, DIMF, true);
    run_pack(stream, eW1 + (size_t)iA * 2 * DIMF * HID + (size_t)DIMF * HID, nullptr, BpL, DIMF, HID);
    run_gemm(stream, xp16, BpL, nullptr, Qbuf, NPOINT, HID, DIMF, true);
    run_pack(stream, eW2 + (size_t)iA * HID * DIMF, nullptr, Bp2, HID, DIMF);
    run_fill(stream, maxbuf, -INFINITY, (long long)NPOINT * DIMF);
    k_edgeconv<<<dim3(EPP / 16), dim3(256), 0, stream>>>(Dbuf, Qbuf, Bp2, eb2 + (size_t)iA * DIMF,
                                                         pp_src, pp_dst, maxbuf, EPP);
    k_finmax<<<dim3((unsigned)(((long long)NPOINT * DIMF + 255) / 256)), dim3(256), 0, stream>>>(
        maxbuf, gb + (size_t)i_lp * DIMF, point_a, (long long)NPOINT * DIMF);
    // GAT lp aggregation (uses HsL, score, db -- untouched by EdgeConv)
    run_aggregate(stream, score, db, HsL, lp_src, lp_dst, point_a, ELP);

    // ================= conv B =================
    float* oL = (layer == NLAYERS - 1) ? out_lego : lego_a;
    float* oP = (layer == NLAYERS - 1) ? out_point : point_a;
    run_cvt(stream, lego_a, xl16, (long long)NLEGO * DIMF);
    run_cvt(stream, point_a, xp16, (long long)NPOINT * DIMF);

    // TransformerConv lego (instance iB)
    run_pack(stream, tWq + (size_t)iB * DIMF * DIMF, nullptr, BpS, DIMF, DIMF);
    run_gemm(stream, xl16, BpS, tbq + (size_t)iB * DIMF, bufQ, NLEGO, DIMF, DIMF, false);
    run_pack(stream, tWk + (size_t)iB * DIMF * DIMF, nullptr, BpS, DIMF, DIMF);
    run_gemm(stream, xl16, BpS, tbk + (size_t)iB * DIMF, bufK, NLEGO, DIMF, DIMF, false);
    run_pack(stream, tWv + (size_t)iB * DIMF * DIMF, nullptr, BpS, DIMF, DIMF);
    run_gemm(stream, xl16, BpS, tbv + (size_t)iB * DIMF, bufV, NLEGO, DIMF, DIMF, false);
    run_pack(stream, tWs + (size_t)iB * DIMF * DIMF, nullptr, BpS, DIMF, DIMF);
    run_gemm(stream, xl16, BpS, tbs + (size_t)iB * DIMF, oL, NLEGO, DIMF, DIMF, false);
    k_trans_score<<<dim3((ELL + 7) / 8), dim3(256), 0, stream>>>(bufQ, bufK, ll_src, ll_dst, score, ELL);
    run_softmax(stream, score, ll_dst, ELL, NLEGO, mb, db);
    run_aggregate(stream, score, db, bufV, ll_src, ll_dst, oL, ELL);

    // EdgeConv point (instance iB)
    run_pack(stream, eW1 + (size_t)iB * 2 * DIMF * HID,
             eW1 + (size_t)iB * 2 * DIMF * HID + (size_t)DIMF * HID, BpL, DIMF, HID);
    run_gemm(stream, xp16, BpL, eb1 + (size_t)iB * HID, Dbuf, NPOINT, HID, DIMF, true);
    run_pack(stream, eW1 + (size_t)iB * 2 * DIMF * HID + (size_t)DIMF * HID, nullptr, BpL, DIMF, HID);
    run_gemm(stream, xp16, BpL, nullptr, Qbuf, NPOINT, HID, DIMF, true);
    run_pack(stream, eW2 + (size_t)iB * HID * DIMF, nullptr, Bp2, HID, DIMF);
    run_fill(stream, maxbuf, -INFINITY, (long long)NPOINT * DIMF);
    k_edgeconv<<<dim3(EPP / 16), dim3(256), 0, stream>>>(Dbuf, Qbuf, Bp2, eb2 + (size_t)iB * DIMF,
                                                         pp_src, pp_dst, maxbuf, EPP);
    k_finmax<<<dim3((unsigned)(((long long)NPOINT * DIMF + 255) / 256)), dim3(256), 0, stream>>>(
        maxbuf, nullptr, oP, (long long)NPOINT * DIMF);

    curL = oL;
    curP = oP;
  }
  (void)in_sizes; (void)n_in; (void)out_size; (void)ws_size;
}
